// JointNetwork_6545530159178
// MI455X (gfx1250) — matrix operand order repaired and checked by reading, not compiled, not run on hardware
//
#include <hip/hip_runtime.h>
#include <cstddef>
#include <cstdint>

// ---------------------------------------------------------------------------
// RNN-T Joint network, fused for MI455X (gfx1250, wave32, WMMA bf16 16x16x32).
//   enc_p  = enc_out @ W_enc + b_enc          (2048 x 512)   -> ws (fp32)
//   pred_p = pred_out @ W_pred + b_pred       ( 512 x 512)   -> ws (fp32)
//   logits = tanh(enc_p[bt] + pred_p[bu]) @ W_joint + b_joint
// The (8,256,64,512) joint tensor (268MB fp32) is never materialized in HBM:
// each block builds one 16-row bf16 A-panel (16x512) in LDS, in the exact
// per-lane WMMA-A layout, and 4 waves consume it against pre-packed bf16 B.
// Addressing uses scalar-base + 32-bit voffset + immediate ioffsets so the
// WMMA inner loop carries ~2 VALU adds of address math per K-step.
// ---------------------------------------------------------------------------

typedef __attribute__((ext_vector_type(16))) __bf16 v16bf;
typedef __attribute__((ext_vector_type(8)))  float  v8f;

#define ENC_DIM   512
#define PRED_DIM  640
#define JOINT_DIM 512
#define VOCAB     1024
#define BATCH     8
#define TT        256
#define UU        64

// ---- workspace layout (bytes) ----
#define OFF_ENCP    ((size_t)0)                       // 2048*512*4 = 4 MB
#define OFF_PREDP   (OFF_ENCP  + (size_t)2048*512*4)  // 512*512*4  = 1 MB
#define OFF_PWENC   (OFF_PREDP + (size_t)512*512*4)   // 512*512*2 bf16
#define OFF_PWPRED  (OFF_PWENC + (size_t)512*512*2)   // 640*512*2 bf16
#define OFF_PWJOINT (OFF_PWPRED + (size_t)640*512*2)  // 512*1024*2 bf16

// ---------------------------------------------------------------------------
// Pack row-major fp32 W[K][N] into bf16 per-lane WMMA B layout:
// for (kc, nt, lane): 16 contiguous bf16 = W[kc*32 + (lane>>4)*8 + (i&7) + (i>>3)*16][nt*16 + (lane&15)]
// so the consumer loads one aligned 32-byte chunk per lane per (kc, n-tile).
// ---------------------------------------------------------------------------
__global__ void pack_w_bf16(const float* __restrict__ W, __bf16* __restrict__ P,
                            int K, int N) {
  const int numNt = N >> 4;
  const int id = blockIdx.x * blockDim.x + threadIdx.x;
  const int total = (K >> 5) * numNt * 32;
  if (id >= total) return;
  const int lane = id & 31;
  const int nt   = (id >> 5) % numNt;
  const int kc   = (id >> 5) / numNt;
  const int n     = nt * 16 + (lane & 15);
  const int kbase = kc * 32 + (lane >> 4) * 8;
  __bf16* dst = P + (size_t)id * 16;
#pragma unroll
  for (int i = 0; i < 16; ++i)
    dst[i] = (__bf16)W[(size_t)(kbase + (i & 7) + ((i >> 3) << 4)) * N + n];
}

__device__ __forceinline__ void load16f(const float* __restrict__ p, int k0,
                                        float v[16]) {
  float4 t0 = *(const float4*)(p + k0);
  float4 t1 = *(const float4*)(p + k0 + 4);
  float4 t2 = *(const float4*)(p + k0 + 16);
  float4 t3 = *(const float4*)(p + k0 + 20);
  v[0]=t0.x;  v[1]=t0.y;  v[2]=t0.z;  v[3]=t0.w;
  v[4]=t1.x;  v[5]=t1.y;  v[6]=t1.z;  v[7]=t1.w;
  v[8]=t2.x;  v[9]=t2.y;  v[10]=t2.z; v[11]=t2.w;
  v[12]=t3.x; v[13]=t3.y; v[14]=t3.z; v[15]=t3.w;
}

// ---------------------------------------------------------------------------
// Generic projection GEMM: Y[M][N] = X[M][K] @ W + bias, via bf16 WMMA,
// fp32 accumulate.  One wave = 16 rows x (NT*16) cols.  blockDim.x = 128.
// ---------------------------------------------------------------------------
template <int NT>
__global__ void proj_wmma(const float* __restrict__ X,
                          const __bf16* __restrict__ PW,
                          const float* __restrict__ bias,
                          float* __restrict__ Y, int M, int K, int N) {
  const int lane  = threadIdx.x & 31;
  const int wave  = threadIdx.x >> 5;
  const int mtile = blockIdx.x * (blockDim.x >> 5) + wave;
  const int ngrp  = blockIdx.y;
  const int numNt = N >> 4;
  const int m  = lane & 15;
  const int hi = lane >> 4;
  const float* xrow = X + (size_t)(mtile * 16 + m) * K;

  // B: scalar base + 32-bit voffset (bytes), bumped once per K-step.
  const char* PB = (const char*)PW;
  uint32_t voff = (uint32_t)((ngrp * NT) * 32 + lane) * 32u;
  const uint32_t vstep = (uint32_t)numNt * 32u * 32u;  // one kc = numNt tiles

  v8f acc[NT] = {};
  const int numKc = K >> 5;
#pragma unroll 1
  for (int kc = 0; kc < numKc; ++kc) {
    const int k0 = kc * 32 + hi * 8;
    float xv[16];
    load16f(xrow, k0, xv);
    v16bf a;
#pragma unroll
    for (int i = 0; i < 16; ++i) a[i] = (__bf16)xv[i];
#pragma unroll
    for (int t = 0; t < NT; ++t) {
      const v16bf bv = *(const v16bf*)(PB + voff + t * 1024);
      acc[t] = __builtin_amdgcn_wmma_f32_16x16x32_bf16(
          false, a, false, bv, (short)0, acc[t], false, false);
    }
    voff += vstep;
  }
  // epilogue: one base, immediate offsets (r rows = N*4 bytes, t tiles = 64B)
  const char* ob = (const char*)(Y + (size_t)(mtile * 16 + hi * 8) * N
                                   + (ngrp * NT) * 16 + m);
  const char* bb = (const char*)(bias + (ngrp * NT) * 16 + m);
#pragma unroll
  for (int t = 0; t < NT; ++t) {
    const float bv = *(const float*)(bb + t * 64);
#pragma unroll
    for (int r = 0; r < 8; ++r)
      *(float*)(ob + (size_t)r * N * 4 + t * 64) = acc[t][r] + bv;
  }
}

// ---------------------------------------------------------------------------
// Fused joint kernel.  One block (128 thr = 4 waves) owns one 16-row m-tile
// (16 consecutive u with fixed (b,t)).  All threads cooperatively build the
// bf16 A-panel  tanh(enc_p + pred_p)  for the full K=512 into LDS, stored in
// the per-lane WMMA-A swizzle so each wave's A fetch per K-step is one
// aligned 32-byte ds_load.  Wave w then accumulates NT=8 n-tiles (128 cols)
// from n-group (blockIdx.y*4 + w).  grid = (8192, 2) -> tanh computed only
// 2x per element instead of 8x.
// ---------------------------------------------------------------------------
__global__ void __launch_bounds__(128)
joint_wmma(const float* __restrict__ encP,
           const float* __restrict__ predP,
           const __bf16* __restrict__ PWj,
           const float* __restrict__ bj,
           float* __restrict__ out) {
  constexpr int NT = 8;
  constexpr int numNt = VOCAB >> 4;       // 64
  __shared__ __bf16 Alds[16 * 32 * 16];   // [kc][lane][16] = 16 KB

  const int tid  = threadIdx.x;
  const int lane = tid & 31;
  const int wave = tid >> 5;

  const int mtile = blockIdx.x;           // 0..8191
  const int flat  = mtile * 16;           // base row (b,t,u0)
  const int u0    = flat & (UU - 1);
  const int bt    = flat >> 6;            // b*T + t
  const int b     = bt >> 8;              // T = 256

  // ---- cooperative A-panel build: 16 rows x 512 K, tanh once per element ----
  {
    const int fm = tid & 15;              // row within tile
    const int fg = tid >> 4;              // 0..7, each covers 64 contiguous K
    const float* erow = encP  + (size_t)bt * JOINT_DIM;
    const float* prow = predP + (size_t)(b * UU + u0 + fm) * JOINT_DIM;
#pragma unroll
    for (int kk = 0; kk < 64; kk += 4) {
      const int k = fg * 64 + kk;
      float4 e = *(const float4*)(erow + k);
      float4 p = *(const float4*)(prow + k);
      float s[4] = {e.x + p.x, e.y + p.y, e.z + p.z, e.w + p.w};
#pragma unroll
      for (int j = 0; j < 4; ++j) {
        const int kj = k + j;
        const int kc = kj >> 5;
        const int c  = kj & 31;
        const int fhi = (c >> 3) & 1;                 // K 8-15 / 24-31 -> hi half
        const int i   = (c & 7) + ((c >> 4) << 3);    // ext-vector element
        const int fl  = fm + fhi * 16;                // owning lane
        Alds[((kc * 32 + fl) << 4) + i] = (__bf16)tanhf(s[j]);
      }
    }
  }
  __syncthreads();

  // ---- WMMA main loop: wave w covers n-tiles [ngrp*8, ngrp*8+8) ----
  const int ngrp = blockIdx.y * 4 + wave;             // 0..7, 128 cols each
  const char* PA = (const char*)Alds;
  uint32_t aoff = (uint32_t)lane * 32u;               // +1024B per kc
  const char* PB = (const char*)PWj;
  uint32_t voff = (uint32_t)((ngrp * NT) * 32 + lane) * 32u;  // +65536B per kc

  v8f acc[NT] = {};
#pragma unroll 1
  for (int kc = 0; kc < JOINT_DIM / 32; ++kc) {
    const v16bf a = *(const v16bf*)(PA + aoff);
#pragma unroll
    for (int t = 0; t < NT; ++t) {
      const v16bf bv = *(const v16bf*)(PB + voff + t * 1024);
      acc[t] = __builtin_amdgcn_wmma_f32_16x16x32_bf16(
          false, a, false, bv, (short)0, acc[t], false, false);
    }
    aoff += 1024u;
    voff += 65536u;
  }

  // ---- epilogue: bias + fp32 store (C/D layout: row = r + 8*hi, col = lane&15)
  const int m  = lane & 15;
  const int hi = lane >> 4;
  const char* ob = (const char*)(out + (size_t)(flat + hi * 8) * VOCAB
                                     + (ngrp * NT) * 16 + m);
  const char* bb = (const char*)(bj + (ngrp * NT) * 16 + m);
#pragma unroll
  for (int t = 0; t < NT; ++t) {
    const float biasv = *(const float*)(bb + t * 64);
#pragma unroll
    for (int r = 0; r < 8; ++r)
      *(float*)(ob + r * 4096 + t * 64) = acc[t][r] + biasv;   // r*VOCAB*4, t*16*4
  }
}

// ---------------------------------------------------------------------------
extern "C" void kernel_launch(void* const* d_in, const int* in_sizes, int n_in,
                              void* d_out, int out_size, void* d_ws,
                              size_t ws_size, hipStream_t stream) {
  (void)in_sizes; (void)n_in; (void)out_size; (void)ws_size;
  const float* enc_out  = (const float*)d_in[0];
  const float* pred_out = (const float*)d_in[1];
  const float* W_enc    = (const float*)d_in[2];
  const float* b_enc    = (const float*)d_in[3];
  const float* W_pred   = (const float*)d_in[4];
  const float* b_pred   = (const float*)d_in[5];
  const float* W_joint  = (const float*)d_in[6];
  const float* b_joint  = (const float*)d_in[7];
  float* out = (float*)d_out;

  char* ws = (char*)d_ws;
  float*  encP    = (float*)(ws + OFF_ENCP);
  float*  predP   = (float*)(ws + OFF_PREDP);
  __bf16* pWenc   = (__bf16*)(ws + OFF_PWENC);
  __bf16* pWpred  = (__bf16*)(ws + OFF_PWPRED);
  __bf16* pWjoint = (__bf16*)(ws + OFF_PWJOINT);

  // 1) pack weights into WMMA-B bf16 layout (tiny, one-shot per call)
  pack_w_bf16<<<64,  256, 0, stream>>>(W_enc,   pWenc,   ENC_DIM,   JOINT_DIM);
  pack_w_bf16<<<80,  256, 0, stream>>>(W_pred,  pWpred,  PRED_DIM,  JOINT_DIM);
  pack_w_bf16<<<128, 256, 0, stream>>>(W_joint, pWjoint, JOINT_DIM, VOCAB);

  // 2) projections (fp32 in/out, bf16 WMMA, fp32 accumulate)
  proj_wmma<4><<<dim3(32, 8), 128, 0, stream>>>(enc_out,  pWenc,  b_enc,
                                                encP, BATCH * TT, ENC_DIM, JOINT_DIM);
  proj_wmma<4><<<dim3(8, 8), 128, 0, stream>>>(pred_out, pWpred, b_pred,
                                               predP, BATCH * UU, PRED_DIM, JOINT_DIM);

  // 3) fused tanh + joint GEMM: one block per m-tile, 2 n-halves
  joint_wmma<<<dim3(8192, 2), 128, 0, stream>>>(encP, predP, pWjoint, b_joint, out);
}
